// GRU_84593675862766
// MI455X (gfx1250) — compile-verified, matrix operand order checked by reading, not run on hardware
//
#include <hip/hip_runtime.h>
#include <hip/hip_bf16.h>

// ---------------------------------------------------------------------------
// Haste GRU (reset-after) for MI455X / gfx1250, wave32 + WMMA bf16 + async LDS.
//   T=1024, N=64, I=256, H=512, 3H=1536
// Prep   : convert x / W^T / R^T to bf16 once (removes all cvt from hot loops)
// Phase 1: Wx = x @ kernel + bias   (WMMA bf16, LDS staged via async copies)
// Phase 2: persistent scan, 16 WGs, R slice resident in LDS (bf16),
//          h ping-ponged as bf16 in global -> per-step staging is a pure
//          global_load_async_to_lds_b128 stream; grid barrier per timestep.
// ---------------------------------------------------------------------------

#define Tt   1024
#define Nb   64
#define Ii   256
#define Hh   512
#define H3   1536
#define NWG  16
#define HS   32          // h-columns per workgroup (Hh / NWG)

typedef __attribute__((ext_vector_type(16))) __bf16 bf16x16;
typedef __attribute__((ext_vector_type(8)))  __bf16 bf16x8;
typedef __attribute__((ext_vector_type(4)))  __bf16 bf16x4;
typedef __attribute__((ext_vector_type(8)))  float  f32x8;

// --- CDNA5 async global->LDS copy (ASYNCcnt), per CDNA5 ISA --------
__device__ __forceinline__ unsigned lds_off(const void* p) {
    // generic shared pointer: low 32 bits are the wave-relative LDS byte addr
    return (unsigned)(unsigned long long)p;
}
__device__ __forceinline__ void async_copy_b128(unsigned lds_addr, const void* g) {
    asm volatile("global_load_async_to_lds_b128 %0, %1, off"
                 :: "v"(lds_addr), "v"((unsigned long long)g) : "memory");
}
__device__ __forceinline__ void wait_async0() {
    asm volatile("s_wait_asynccnt 0x0" ::: "memory");
}

// Load one 16x32 bf16 WMMA operand fragment for this lane from an LDS row.
// ISA layout: lane half h owns K = h*8+[0..7] and K = 16+h*8+[0..7]
// -> two contiguous 16-byte chunks -> two ds_load_b128.
__device__ __forceinline__ bf16x16 load_frag(const __bf16* row, int koff) {
    bf16x8 lo = *(const bf16x8*)(row + koff);
    bf16x8 hi = *(const bf16x8*)(row + koff + 16);
    return __builtin_shufflevector(lo, hi, 0, 1, 2, 3, 4, 5, 6, 7,
                                           8, 9, 10, 11, 12, 13, 14, 15);
}
__device__ __forceinline__ f32x8 zero8() {
    f32x8 z = {0.f, 0.f, 0.f, 0.f, 0.f, 0.f, 0.f, 0.f};
    return z;
}

// ---------------------------------------------------------------------------
// Prep kernels: one-shot f32 -> bf16 conversions (and transposes for B-side)
// ---------------------------------------------------------------------------
__global__ __launch_bounds__(256) void cvt_x_kernel(const float* __restrict__ x,
                                                    __bf16* __restrict__ xb) {
    size_t i = ((size_t)blockIdx.x * 256 + threadIdx.x) * 4;  // 16,777,216 elems
    float4 v = *(const float4*)(x + i);
    bf16x4 o = {(__bf16)v.x, (__bf16)v.y, (__bf16)v.z, (__bf16)v.w};
    *(bf16x4*)(xb + i) = o;
}
// Wtb[n][k] = W[k][n], bf16, n<1536, k<256
__global__ __launch_bounds__(256) void cvt_wt_kernel(const float* __restrict__ w,
                                                     __bf16* __restrict__ wtb) {
    int k = blockIdx.x;                       // 256 blocks
    for (int i = 0; i < 6; ++i) {
        int n = i * 256 + threadIdx.x;        // coalesced read along n
        wtb[(size_t)n * Ii + k] = (__bf16)w[(size_t)k * H3 + n];
    }
}
// Rtb[n][k] = R[k][n], bf16, n<1536, k<512
__global__ __launch_bounds__(256) void cvt_rt_kernel(const float* __restrict__ r,
                                                     __bf16* __restrict__ rtb) {
    int k = blockIdx.x;                       // 512 blocks
    for (int i = 0; i < 6; ++i) {
        int n = i * 256 + threadIdx.x;
        rtb[(size_t)n * Hh + k] = (__bf16)r[(size_t)k * H3 + n];
    }
}

__global__ void gru_init_kernel(float* h0, __bf16* hbf0, unsigned* bar) {
    int i = blockIdx.x * blockDim.x + threadIdx.x;   // 32768 threads
    if (i < Nb * Hh) { h0[i] = 0.f; hbf0[i] = (__bf16)0.f; }
    if (i == 0) *bar = 0u;
}

// ---------------------------------------------------------------------------
// Phase 1: Wx = x @ kernel + bias. Block 64(M) x 128(N); 8 waves = 4x2;
// wave strip 16x64 (4 accumulators). K staged 32/iter via async LDS copies.
// ---------------------------------------------------------------------------
__global__ __launch_bounds__(256) void wx_gemm_kernel(
    const __bf16* __restrict__ xb, const __bf16* __restrict__ wtb,
    const float* __restrict__ bias, float* __restrict__ wx) {
    __shared__ __bf16 sA[64 * 40];    // [64 rows][K 32, pad->40]
    __shared__ __bf16 sB[128 * 40];   // [128 cols][K 32, pad->40]

    const int tid  = threadIdx.x;
    const int wid  = tid >> 5;
    const int lane = tid & 31;
    const int lr   = lane & 15;
    const int hh   = lane >> 4;
    const int wm   = wid & 3;
    const int wn   = wid >> 2;
    const int mblk = blockIdx.x * 64;
    const int nblk = blockIdx.y * 128;

    f32x8 acc[4] = {zero8(), zero8(), zero8(), zero8()};

    for (int k0 = 0; k0 < Ii; k0 += 32) {
        __syncthreads();
        // A: 64 rows x 32 K (1 x b128 per thread)
        {
            int r  = tid >> 2;
            int kc = (tid & 3) * 8;
            async_copy_b128(lds_off(&sA[r * 40 + kc]),
                            xb + (size_t)(mblk + r) * Ii + k0 + kc);
        }
        // B: 128 cols x 32 K from Wtb[n][256] (2 x b128 per thread)
#pragma unroll
        for (int i = 0; i < 2; ++i) {
            int linear = i * 256 + tid;       // 0..511
            int nn = linear >> 2;             // 0..127
            int kc = (linear & 3) * 8;
            async_copy_b128(lds_off(&sB[nn * 40 + kc]),
                            wtb + (size_t)(nblk + nn) * Ii + k0 + kc);
        }
        wait_async0();
        __syncthreads();

        bf16x16 afrag = load_frag(&sA[(wm * 16 + lr) * 40], hh * 8);
#pragma unroll
        for (int nt = 0; nt < 4; ++nt) {
            bf16x16 bfrag = load_frag(&sB[(wn * 64 + nt * 16 + lr) * 40], hh * 8);
            acc[nt] = __builtin_amdgcn_wmma_f32_16x16x32_bf16(
                false, afrag, false, bfrag, (short)0, acc[nt], false, false);
        }
    }

#pragma unroll
    for (int nt = 0; nt < 4; ++nt) {
        int col = nblk + wn * 64 + nt * 16 + lr;
        float b = bias[col];
#pragma unroll
        for (int e = 0; e < 8; ++e) {
            int row = mblk + wm * 16 + e + 8 * hh;
            wx[(size_t)row * H3 + col] = acc[nt][e] + b;
        }
    }
}

// ---------------------------------------------------------------------------
// Phase 2: persistent scan. WG b owns h-cols [b*32, b*32+32).
// LDS: sR [96 cols][520] bf16 (z|r|g) + sH [64][520] bf16.
// Wave: rows wm*16..+16, 16-col half wn; fuses z/r/g tiles in registers.
// ---------------------------------------------------------------------------
__global__ __launch_bounds__(256) void gru_scan_kernel(
    const float* __restrict__ wx, const __bf16* __restrict__ rtb,
    const float* __restrict__ rbias, const float* __restrict__ h0,
    __bf16* __restrict__ hbf, float* __restrict__ out,
    unsigned* __restrict__ bar) {
    extern __shared__ __bf16 smem[];
    __bf16* sR = smem;                 // [96][520]
    __bf16* sH = smem + 96 * 520;      // [64][520]

    const int tid  = threadIdx.x;
    const int wid  = tid >> 5;
    const int lane = tid & 31;
    const int lr   = lane & 15;
    const int hh   = lane >> 4;
    const int wm   = wid & 3;
    const int wn   = wid >> 2;
    const int j0   = blockIdx.x * HS;

    // preload R slice: 96 cols x 512 K, pure async b128 stream (24 per thread)
    for (int i = 0; i < 24; ++i) {
        int linear = i * 256 + tid;    // 0..6143
        int c  = linear >> 6;          // 0..95
        int ch = linear & 63;          // 16B chunk within row
        int gate = c >> 5, cc = c & 31;
        async_copy_b128(lds_off(&sR[c * 520 + ch * 8]),
                        rtb + (size_t)(gate * Hh + j0 + cc) * Hh + ch * 8);
    }
    wait_async0();

    const int jc = wn * 16 + lr;       // local h-column 0..31
    const int j  = j0 + jc;            // global h-column
    const float rbz = rbias[j];
    const float rbr = rbias[Hh + j];
    const float rbg = rbias[2 * Hh + j];

    for (int t = 0; t < Tt; ++t) {
        const float*  hprev = (t == 0) ? h0 : (out + (size_t)(t - 1) * (Nb * Hh));
        const __bf16* hbp   = hbf + (size_t)(t & 1) * (Nb * Hh);
        __bf16*       hbn   = hbf + (size_t)((t + 1) & 1) * (Nb * Hh);
        const float*  wxt   = wx + (size_t)t * Nb * H3;

        __syncthreads();               // sR/sH free for reuse
        // stage h_prev (bf16) 64x512: pure async b128 stream (16 per thread)
#pragma unroll 4
        for (int i = 0; i < 16; ++i) {
            int linear = i * 256 + tid;  // 0..4095
            int r  = linear >> 6;        // row 0..63
            int ch = linear & 63;
            async_copy_b128(lds_off(&sH[r * 520 + ch * 8]),
                            hbp + (size_t)r * Hh + ch * 8);
        }
        // prefetch the Wx gate rows this lane's epilogue will read
        __builtin_prefetch(wxt + (wm * 16 + 8 * hh) * H3 + j, 0, 1);
        __builtin_prefetch(wxt + (wm * 16 + 8 * hh) * H3 + Hh + j, 0, 1);
        __builtin_prefetch(wxt + (wm * 16 + 8 * hh) * H3 + 2 * Hh + j, 0, 1);
        wait_async0();
        __syncthreads();

        f32x8 accz = zero8(), accr = zero8(), accg = zero8();
        for (int k0 = 0; k0 < Hh; k0 += 32) {
            bf16x16 a  = load_frag(&sH[(wm * 16 + lr) * 520 + k0], hh * 8);
            bf16x16 bz = load_frag(&sR[(wn * 16 + lr) * 520 + k0], hh * 8);
            bf16x16 br = load_frag(&sR[(32 + wn * 16 + lr) * 520 + k0], hh * 8);
            bf16x16 bg = load_frag(&sR[(64 + wn * 16 + lr) * 520 + k0], hh * 8);
            accz = __builtin_amdgcn_wmma_f32_16x16x32_bf16(false, a, false, bz,
                                                           (short)0, accz, false, false);
            accr = __builtin_amdgcn_wmma_f32_16x16x32_bf16(false, a, false, br,
                                                           (short)0, accr, false, false);
            accg = __builtin_amdgcn_wmma_f32_16x16x32_bf16(false, a, false, bg,
                                                           (short)0, accg, false, false);
        }

        float* outt = out + (size_t)t * (Nb * Hh);
#pragma unroll
        for (int e = 0; e < 8; ++e) {
            int m = wm * 16 + e + 8 * hh;          // batch row
            float wz = wxt[m * H3 + j];
            float wr = wxt[m * H3 + Hh + j];
            float wg = wxt[m * H3 + 2 * Hh + j];
            float hp = hprev[m * Hh + j];
            float z = 1.f / (1.f + __expf(-(wz + accz[e] + rbz)));
            float r = 1.f / (1.f + __expf(-(wr + accr[e] + rbr)));
            float g = tanhf(wg + r * (accg[e] + rbg));
            float hn = z * hp + (1.f - z) * g;
            outt[m * Hh + j] = hn;
            hbn[m * Hh + j] = (__bf16)hn;          // bf16 copy for next step's A
            if (t == Tt - 1)
                out[(size_t)Tt * Nb * Hh + m * Hh + j] = hn;   // state = hs[-1]
        }

        // grid barrier: release stores, acquire-spin (dev-scope inv)
        __threadfence();
        __syncthreads();
        if (tid == 0) {
            atomicAdd(bar, 1u);
            unsigned target = (unsigned)(t + 1) * NWG;
            while (__hip_atomic_load(bar, __ATOMIC_ACQUIRE,
                                     __HIP_MEMORY_SCOPE_AGENT) < target) {
                __builtin_amdgcn_s_sleep(2);
            }
        }
        __syncthreads();
    }
}

extern "C" void kernel_launch(void* const* d_in, const int* in_sizes, int n_in,
                              void* d_out, int out_size, void* d_ws, size_t ws_size,
                              hipStream_t stream) {
    const float* x     = (const float*)d_in[0];
    const float* w     = (const float*)d_in[1];
    const float* rker  = (const float*)d_in[2];
    const float* bias  = (const float*)d_in[3];
    const float* rbias = (const float*)d_in[4];
    float* out = (float*)d_out;

    // ws layout (bytes):
    char* base = (char*)d_ws;
    size_t off = 0;
    float*    wxbuf = (float*)(base + off);  off += (size_t)Tt * Nb * H3 * 4;   // 402,653,184
    float*    h0    = (float*)(base + off);  off += (size_t)Nb * Hh * 4;        // 131,072
    unsigned* bar   = (unsigned*)(base + off); off += 64;
    __bf16*   xb    = (__bf16*)(base + off); off += (size_t)Tt * Nb * Ii * 2;   // 33,554,432
    __bf16*   wtb   = (__bf16*)(base + off); off += (size_t)H3 * Ii * 2;        // 786,432
    __bf16*   rtb   = (__bf16*)(base + off); off += (size_t)H3 * Hh * 2;        // 1,572,864
    __bf16*   hbf   = (__bf16*)(base + off); off += (size_t)2 * Nb * Hh * 2;    // 131,072

    gru_init_kernel<<<(Nb * Hh + 255) / 256, 256, 0, stream>>>(h0, hbf, bar);
    cvt_x_kernel<<<(Tt * Nb * Ii / 4) / 256, 256, 0, stream>>>(x, xb);
    cvt_wt_kernel<<<Ii, 256, 0, stream>>>(w, wtb);
    cvt_rt_kernel<<<Hh, 256, 0, stream>>>(rker, rtb);

    dim3 g1(Tt * Nb / 64, H3 / 128);
    wx_gemm_kernel<<<g1, 256, 0, stream>>>(xb, wtb, bias, wxbuf);

    size_t smem_bytes = (size_t)(96 * 520 + 64 * 520) * 2;   // 166,400 B < 320 KB
    gru_scan_kernel<<<NWG, 256, smem_bytes, stream>>>(wxbuf, rtb, rbias, h0, hbf,
                                                      out, bar);
}
